// AttnBlock_43954695307886
// MI455X (gfx1250) — compile-verified, matrix operand order checked by reading, not run on hardware
//
#include <hip/hip_runtime.h>
#include <hip/hip_bf16.h>

// ---------- types ----------
typedef __attribute__((ext_vector_type(16))) __bf16 v16bf;
typedef __attribute__((ext_vector_type(8)))  __bf16 v8bf;
typedef __attribute__((ext_vector_type(8)))  float  v8f;
typedef int v4i_ __attribute__((vector_size(16)));   // matches builtin's b128 pointee

union V16u { v16bf v; v8bf h[2]; };

#define BB 8
#define CC 512
#define NN 4096
#define GG 32
#define CGc 16
#define GN_EPS 1e-5f

// CDNA5 async global->LDS copy path (ASYNCcnt-tracked), with safe fallback.
#if defined(__HIP_DEVICE_COMPILE__) && \
    __has_builtin(__builtin_amdgcn_global_load_async_to_lds_b128) && \
    __has_builtin(__builtin_amdgcn_s_wait_asynccnt)
#define USE_ASYNC_LDS 1
#else
#define USE_ASYNC_LDS 0
#endif

// copy 16 bytes (8 bf16) from global to LDS
__device__ __forceinline__ void copy16(__bf16* lds, const __bf16* g)
{
#if USE_ASYNC_LDS
    __builtin_amdgcn_global_load_async_to_lds_b128(
        (__attribute__((address_space(1))) v4i_*)(g),
        (__attribute__((address_space(3))) v4i_*)(lds),
        0, 0);
#else
    *(uint4*)lds = *(const uint4*)g;
#endif
}

__device__ __forceinline__ void wait_copies()
{
#if USE_ASYNC_LDS
    __builtin_amdgcn_s_wait_asynccnt(0);
#endif
    // sync fallback: __syncthreads() already orders ds_store via DScnt
}

// =====================================================================
// GroupNorm -> bf16, transposed layout hT[b][n][c]  (c contiguous)
// =====================================================================
__global__ __launch_bounds__(256) void gn_kernel(const float* __restrict__ x,
                                                 const float* __restrict__ gsc,
                                                 const float* __restrict__ gbi,
                                                 __bf16* __restrict__ hT)
{
    const int b = blockIdx.x / GG;
    const int g = blockIdx.x % GG;
    const float* xp = x + ((size_t)b * CC + (size_t)g * CGc) * NN;
    const int t = threadIdx.x;

    float s = 0.f, sq = 0.f;
    for (int i = t; i < CGc * NN; i += 256) {
        float v = xp[i];
        s += v; sq += v * v;
    }
    __shared__ float rs_[256], rq_[256];
    rs_[t] = s; rq_[t] = sq;
    __syncthreads();
    for (int st = 128; st > 0; st >>= 1) {
        if (t < st) { rs_[t] += rs_[t + st]; rq_[t] += rq_[t + st]; }
        __syncthreads();
    }
    const float inv_n = 1.0f / (float)(CGc * NN);
    const float mu  = rs_[0] * inv_n;
    const float var = rq_[0] * inv_n - mu * mu;
    const float rsd = rsqrtf(var + GN_EPS);

    for (int i = t; i < CGc * NN; i += 256) {
        int cl = i >> 12;
        int n  = i & (NN - 1);
        int c  = g * CGc + cl;
        float v = (xp[i] - mu) * rsd * gsc[c] + gbi[c];
        hT[((size_t)b * NN + n) * CC + c] = (__bf16)v;
    }
}

// =====================================================================
// fp32 weight -> bf16
// =====================================================================
__global__ __launch_bounds__(256) void w2bf_kernel(const float* __restrict__ in,
                                                   __bf16* __restrict__ out, int n)
{
    int i = blockIdx.x * 256 + threadIdx.x;
    if (i < n) out[i] = (__bf16)in[i];
}

// =====================================================================
// Generic WMMA GEMM:  D[o,n] = scale * sum_k A[o,k] * Bt[n,k]  (+epilogue)
//   A : [M x K] bf16 row-major, ldA;  Bt: [Nn x K] bf16 row-major, ldB
// Block tile 128x128, 8 waves (4 M x 2 N), wave tile 32x64, K-step 32.
// Double-buffered LDS with async global->LDS copies (1 barrier / k-step).
// MODE 0: outB[n*ldOut + o] = bf16(val + bias)      (transposed: q,k)
// MODE 1: outB[o*ldOut + n] = bf16(val + bias)      (natural: v, P@V)
// MODE 2: outF[o*ldOut + n] = val                   (fp32 scores, scaled)
// MODE 3: outF[o*ldOut + n] = val + bias + resid    (final + residual)
// =====================================================================
template<int MODE>
__global__ __launch_bounds__(256) void gemm_tn(const __bf16* __restrict__ A,
                                               const __bf16* __restrict__ Bt,
                                               const float*  __restrict__ bias,
                                               const float*  __restrict__ resid,
                                               float*        __restrict__ outF,
                                               __bf16*       __restrict__ outB,
                                               int K, int ldA, int ldB, int ldOut,
                                               float scale)
{
    __shared__ __align__(16) __bf16 As[2][128][32];
    __shared__ __align__(16) __bf16 Bs[2][128][32];

    const int tid  = threadIdx.x;
    const int lane = tid & 31;
    const int wave = tid >> 5;
    const int wm   = wave & 3;   // wave M offset: wm*32
    const int wn   = wave >> 2;  // wave N offset: wn*64
    const int lh   = lane & 15;
    const int khalf = (lane < 16) ? 0 : 8;

    const long Mbase = (long)blockIdx.y * 128;
    const long Nbase = (long)blockIdx.x * 128;

    // per-thread cooperative-copy coordinates (two 16B chunks per matrix)
    const int e0 = tid * 8;
    const int r0 = e0 >> 5, c0 = e0 & 31;       // rows 0..63
    const int e1 = (tid + 256) * 8;
    const int r1 = e1 >> 5, c1 = e1 & 31;       // rows 64..127

    auto issue = [&](int buf, int k0) {
        copy16(&As[buf][r0][c0], &A [(Mbase + r0) * (long)ldA + k0 + c0]);
        copy16(&As[buf][r1][c1], &A [(Mbase + r1) * (long)ldA + k0 + c1]);
        copy16(&Bs[buf][r0][c0], &Bt[(Nbase + r0) * (long)ldB + k0 + c0]);
        copy16(&Bs[buf][r1][c1], &Bt[(Nbase + r1) * (long)ldB + k0 + c1]);
    };

    v8f acc[2][4] = {};

    issue(0, 0);
    const int KT = K >> 5;
    for (int kt = 0; kt < KT; ++kt) {
        const int cur = kt & 1;
        wait_copies();
        __syncthreads();
        if (kt + 1 < KT) issue(cur ^ 1, (kt + 1) * 32);

        V16u af[2], bf[4];
        #pragma unroll
        for (int mi = 0; mi < 2; ++mi) {
            int row = wm * 32 + mi * 16 + lh;
            af[mi].h[0] = *(const v8bf*)&As[cur][row][khalf];
            af[mi].h[1] = *(const v8bf*)&As[cur][row][khalf + 16];
        }
        #pragma unroll
        for (int ni = 0; ni < 4; ++ni) {
            int col = wn * 64 + ni * 16 + lh;
            bf[ni].h[0] = *(const v8bf*)&Bs[cur][col][khalf];
            bf[ni].h[1] = *(const v8bf*)&Bs[cur][col][khalf + 16];
        }
        #pragma unroll
        for (int mi = 0; mi < 2; ++mi)
            #pragma unroll
            for (int ni = 0; ni < 4; ++ni)
                acc[mi][ni] = __builtin_amdgcn_wmma_f32_16x16x32_bf16(
                    false, af[mi].v, false, bf[ni].v,
                    (short)0, acc[mi][ni], false, false);
    }

    // epilogue; D tile: VGPR r -> row = r + (lane<16?0:8), col = lane%16
    const int rofs = (lane < 16) ? 0 : 8;
    #pragma unroll
    for (int mi = 0; mi < 2; ++mi) {
        #pragma unroll
        for (int ni = 0; ni < 4; ++ni) {
            #pragma unroll
            for (int r = 0; r < 8; ++r) {
                long o = Mbase + wm * 32 + mi * 16 + r + rofs;
                long n = Nbase + wn * 64 + ni * 16 + lh;
                float val = acc[mi][ni][r] * scale;
                if (MODE == 0) {
                    val += bias ? bias[o] : 0.0f;
                    outB[n * (long)ldOut + o] = (__bf16)val;
                } else if (MODE == 1) {
                    val += bias ? bias[o] : 0.0f;
                    outB[o * (long)ldOut + n] = (__bf16)val;
                } else if (MODE == 2) {
                    outF[o * (long)ldOut + n] = val;
                } else { // MODE == 3
                    outF[o * (long)ldOut + n] = val + bias[o] + resid[o * (long)ldOut + n];
                }
            }
        }
    }
}

// =====================================================================
// Row softmax: S fp32 [4096 x 4096] -> P bf16. One block per row.
// =====================================================================
__global__ __launch_bounds__(256) void softmax_kernel(const float* __restrict__ S,
                                                      __bf16* __restrict__ P)
{
    const long row = blockIdx.x;
    const float* sp = S + row * (long)NN;
    __bf16* pp = P + row * (long)NN;
    const int t = threadIdx.x;

    float v[16];
    float mx = -3.4e38f;
    #pragma unroll
    for (int i = 0; i < 16; ++i) {
        v[i] = sp[t + i * 256];
        mx = fmaxf(mx, v[i]);
    }
    __shared__ float red[256];
    red[t] = mx; __syncthreads();
    for (int st = 128; st > 0; st >>= 1) {
        if (t < st) red[t] = fmaxf(red[t], red[t + st]);
        __syncthreads();
    }
    mx = red[0];
    __syncthreads();

    float sum = 0.f;
    #pragma unroll
    for (int i = 0; i < 16; ++i) { v[i] = __expf(v[i] - mx); sum += v[i]; }
    red[t] = sum; __syncthreads();
    for (int st = 128; st > 0; st >>= 1) {
        if (t < st) red[t] += red[t + st];
        __syncthreads();
    }
    const float inv = 1.0f / red[0];
    #pragma unroll
    for (int i = 0; i < 16; ++i)
        pp[t + i * 256] = (__bf16)(v[i] * inv);
}

// =====================================================================
// launch
// =====================================================================
extern "C" void kernel_launch(void* const* d_in, const int* in_sizes, int n_in,
                              void* d_out, int out_size, void* d_ws, size_t ws_size,
                              hipStream_t stream)
{
    const float* x   = (const float*)d_in[0];
    const float* gsc = (const float*)d_in[1];
    const float* gbi = (const float*)d_in[2];
    const float* wq  = (const float*)d_in[3];
    const float* bq  = (const float*)d_in[4];
    const float* wk  = (const float*)d_in[5];
    const float* bk  = (const float*)d_in[6];
    const float* wv  = (const float*)d_in[7];
    const float* bv  = (const float*)d_in[8];
    const float* wo  = (const float*)d_in[9];
    const float* bo  = (const float*)d_in[10];
    float* out = (float*)d_out;

    char* ws = (char*)d_ws;
    const size_t szBF = (size_t)BB * NN * CC * sizeof(__bf16);   // 33.5 MB
    __bf16* hT  = (__bf16*)(ws);                 // h transposed [B,N,C]; reused as P
    __bf16* qT  = (__bf16*)(ws + 1 * szBF);      // [B,N,C]
    __bf16* kT  = (__bf16*)(ws + 2 * szBF);      // [B,N,C]
    __bf16* vB  = (__bf16*)(ws + 3 * szBF);      // [C, B*N]
    __bf16* oT  = (__bf16*)(ws + 4 * szBF);      // [B,N,C]
    float*  S   = (float*)(ws + 5 * szBF);       // [N,N] fp32 scores (reused)
    __bf16* wqb = (__bf16*)(ws + 5 * szBF + (size_t)NN * NN * sizeof(float));
    __bf16* wkb = wqb + (size_t)CC * CC;
    __bf16* wvb = wkb + (size_t)CC * CC;
    __bf16* wob = wvb + (size_t)CC * CC;

    // 1) GroupNorm -> bf16 transposed
    gn_kernel<<<BB * GG, 256, 0, stream>>>(x, gsc, gbi, hT);

    // 2) weights -> bf16
    const int WN = CC * CC;
    w2bf_kernel<<<WN / 256, 256, 0, stream>>>(wq, wqb, WN);
    w2bf_kernel<<<WN / 256, 256, 0, stream>>>(wk, wkb, WN);
    w2bf_kernel<<<WN / 256, 256, 0, stream>>>(wv, wvb, WN);
    w2bf_kernel<<<WN / 256, 256, 0, stream>>>(wo, wob, WN);

    // 3) projections over all batches at once (columns = B*N = 32768)
    dim3 gProj(BB * NN / 128, CC / 128);  // (256, 4)
    gemm_tn<0><<<gProj, 256, 0, stream>>>(wqb, hT, bq, nullptr, nullptr, qT,
                                          CC, CC, CC, CC, 1.0f);
    gemm_tn<0><<<gProj, 256, 0, stream>>>(wkb, hT, bk, nullptr, nullptr, kT,
                                          CC, CC, CC, CC, 1.0f);
    gemm_tn<1><<<gProj, 256, 0, stream>>>(wvb, hT, bv, nullptr, nullptr, vB,
                                          CC, CC, CC, BB * NN, 1.0f);

    // 4) attention per batch (S and P buffers reused; stream serializes)
    const float sc = 0.044194173824159216f;  // 512^-0.5
    for (int b = 0; b < BB; ++b) {
        gemm_tn<2><<<dim3(NN / 128, NN / 128), 256, 0, stream>>>(
            qT + (size_t)b * NN * CC, kT + (size_t)b * NN * CC,
            nullptr, nullptr, S, nullptr, CC, CC, CC, NN, sc);
        softmax_kernel<<<NN, 256, 0, stream>>>(S, hT);
        gemm_tn<1><<<dim3(CC / 128, NN / 128), 256, 0, stream>>>(
            hT, vB + (size_t)b * NN, nullptr, nullptr, nullptr,
            oT + (size_t)b * NN * CC, NN, NN, BB * NN, CC, 1.0f);
        gemm_tn<3><<<dim3(NN / 128, CC / 128), 256, 0, stream>>>(
            wob, oT + (size_t)b * NN * CC, bo, x + (size_t)b * CC * NN,
            out + (size_t)b * CC * NN, nullptr, CC, CC, CC, NN, 1.0f);
    }
}
